// XFeat_36172214567510
// MI455X (gfx1250) — compile-verified, matrix operand order checked by reading, not run on hardware
//
#include <hip/hip_runtime.h>
#include <hip/hip_bf16.h>
#include <stdint.h>

typedef _Float16 v16h __attribute__((ext_vector_type(16)));
typedef float    v8f  __attribute__((ext_vector_type(8)));
typedef unsigned u32x4 __attribute__((ext_vector_type(4)));
typedef unsigned u32x8 __attribute__((ext_vector_type(8)));

#define BATCH 8
#define HC 160
#define WC 160
#define HFULL 1280
#define WFULL 1280
#define NPIX (HC * WC)            // 25600 coarse cells per batch
#define TOPK 4096
#define CAP 65536                 // candidate budget per batch
#define THR 0.05f
#define GUARDF 4096               // guard floats around heatmap

// ---------------------------------------------------------------------------
// Kernel 0: init counters + guard bands (deterministic halo reads for TDM)
// ---------------------------------------------------------------------------
__global__ void k_init(unsigned* __restrict__ cnt,
                       float* __restrict__ g1, float* __restrict__ g2) {
  int t = threadIdx.x;
  if (t < BATCH) cnt[t] = 0u;
  for (int i = t; i < GUARDF; i += blockDim.x) { g1[i] = 0.f; g2[i] = 0.f; }
}

// ---------------------------------------------------------------------------
// Kernel 1: softmax over 65 logits (keep 64) + 8x8 pixel shuffle -> heatmap
// One thread per coarse cell; channel loads are coalesced across lanes.
// ---------------------------------------------------------------------------
__global__ void k_heatmap(const float* __restrict__ K1, float* __restrict__ K1h) {
  int idx = blockIdx.x * blockDim.x + threadIdx.x;
  if (idx >= BATCH * NPIX) return;
  int b = idx / NPIX, cell = idx - b * NPIX;
  int cy = cell / WC, cx = cell - cy * WC;
  const float* src = K1 + (size_t)b * 65 * NPIX + cell;
  float v[65];
  float mx = -1e30f;
#pragma unroll
  for (int c = 0; c < 65; ++c) { v[c] = src[(size_t)c * NPIX]; mx = fmaxf(mx, v[c]); }
  float s = 0.f;
#pragma unroll
  for (int c = 0; c < 65; ++c) { v[c] = __expf(v[c] - mx); s += v[c]; }
  float inv = 1.f / s;
  float* dst = K1h + (size_t)b * HFULL * WFULL;
#pragma unroll
  for (int c = 0; c < 64; ++c) {
    int oy = cy * 8 + (c >> 3), ox = cx * 8 + (c & 7);
    dst[(size_t)oy * WFULL + ox] = v[c] * inv;
  }
}

// ---------------------------------------------------------------------------
// Kernel 2: per-pixel inverse L2 norm over 64 channels of M1, via WMMA.
// Each wave handles 16 pixels: A = squared values (16x64 as two 16x32 f16
// tiles), B = ones => accumulator rows hold channel sums.
// ---------------------------------------------------------------------------
__global__ void k_invnorm(const float* __restrict__ M1, float* __restrict__ invn) {
  int wave = blockIdx.x * (blockDim.x >> 5) + (threadIdx.x >> 5);
  int lane = threadIdx.x & 31;
  int pixBase = wave * 16;               // NPIX divisible by 16, no straddle
  if (pixBase >= BATCH * NPIX) return;
  int b = pixBase / NPIX, p = pixBase - b * NPIX;
  const float* base = M1 + (size_t)b * 64 * NPIX + p;
  int m = lane & 15;                     // pixel row of the A matrix
  int khalf = (lane >= 16) ? 8 : 0;      // lane-group K offset per ISA layout

  v16h ones;
#pragma unroll
  for (int i = 0; i < 16; ++i) ones[i] = (_Float16)1.0f;

  v8f acc = {};
#pragma unroll
  for (int s = 0; s < 2; ++s) {          // two K=32 chunks cover 64 channels
    v16h a;
#pragma unroll
    for (int i = 0; i < 16; ++i) {
      int ch = s * 32 + ((i < 8) ? i : i + 8) + khalf;  // 16-bit A VGPR layout
      float x = base[(size_t)ch * NPIX + m];
      a[i] = (_Float16)(x * x);
    }
    acc = __builtin_amdgcn_wmma_f32_16x16x32_f16(false, a, false, ones,
                                                 (short)0, acc, false, false);
  }
  // D layout: lanes 0-15 VGPR r = row r (pixels 0..7); lanes 16-31 = rows 8..15
  if (lane == 0) {
#pragma unroll
    for (int r = 0; r < 8; ++r)
      invn[(size_t)b * NPIX + p + r] = 1.f / fmaxf(sqrtf(acc[r]), 1e-12f);
  } else if (lane == 16) {
#pragma unroll
    for (int r = 0; r < 8; ++r)
      invn[(size_t)b * NPIX + p + 8 + r] = 1.f / fmaxf(sqrtf(acc[r]), 1e-12f);
  }
}

// ---------------------------------------------------------------------------
// Kernel 3: 5x5 NMS on the heatmap. A 36x36 tile (32x32 + halo) is staged
// into LDS by the Tensor Data Mover; peaks get scored (nearest heatmap x
// bilinear reliability) and compacted into a per-batch candidate list.
// ---------------------------------------------------------------------------
__global__ void k_nms(const float* __restrict__ K1h, const float* __restrict__ H1,
                      unsigned* __restrict__ cnt, uint2* __restrict__ cand) {
  __shared__ float sh[36 * 36];
  int bx = blockIdx.x % 40;
  int by = (blockIdx.x / 40) % 40;
  int b  = blockIdx.x / 1600;
  int x0 = bx * 32, y0 = by * 32;
  const float* hm = K1h + (size_t)b * HFULL * WFULL;

  if (threadIdx.x < 32) {                 // wave 0 issues the TDM descriptor
    uint64_t gaddr = (uint64_t)(uintptr_t)(hm + (long)(y0 - 2) * WFULL + (x0 - 2));
    unsigned ldsOff = (unsigned)(uintptr_t)(&sh[0]);
    u32x4 g0;
    g0.x = 1u;                            // count=1, user descriptor
    g0.y = ldsOff;                        // lds_addr
    g0.z = (unsigned)gaddr;               // global_addr[31:0]
    g0.w = (unsigned)((gaddr >> 32) & 0x1FFFFFFu) | (2u << 30);  // addr hi | type=2
    u32x8 g1;
    g1[0] = 2u << 16;                     // data_size=4B, mask=0, flags=0
    g1[1] = 0u;                           // tensor_dim0 low16 = 0 (dim=1<<30)
    g1[2] = 0x4000u;                      // tensor_dim0 hi | tensor_dim1 low16
    g1[3] = 0x4000u | (36u << 16);        // tensor_dim1 hi | tile_dim0=36
    g1[4] = 36u;                          // tile_dim1=36, tile_dim2=0
    g1[5] = (unsigned)WFULL;              // tensor_dim0_stride = 1280
    g1[6] = 0u;
    g1[7] = 0u;
    asm volatile("tensor_load_to_lds %0, %1" :: "s"(g0), "s"(g1) : "memory");
    __builtin_amdgcn_s_wait_tensorcnt(0);
  }
  __syncthreads();

  int lx = threadIdx.x & 31;
#pragma unroll
  for (int r = 0; r < 4; ++r) {
    int ly = (threadIdx.x >> 5) + r * 8;
    int gx = x0 + lx, gy = y0 + ly;
    float c = sh[(ly + 2) * 36 + (lx + 2)];
    if (c > THR) {
      float mx = -1e30f;
#pragma unroll
      for (int dy = 0; dy < 5; ++dy)
#pragma unroll
        for (int dx = 0; dx < 5; ++dx)
          mx = fmaxf(mx, sh[(ly + dy) * 36 + (lx + dx)]);
      if (c == mx) {
        float snear = (gx == (WFULL - 1) || gy == (HFULL - 1)) ? 0.f : c;
        // bilinear sample of H1 (160x160), align_corners=False unnormalize
        float u = gx * (160.f / 1279.f) - 0.5f;
        float v = gy * (160.f / 1279.f) - 0.5f;
        int xf = (int)floorf(u), yf = (int)floorf(v);
        float tx = u - xf, ty = v - yf;
        const float* h = H1 + (size_t)b * NPIX;
        auto g = [&](int yy, int xx) -> float {
          return (yy >= 0 && yy < HC && xx >= 0 && xx < WC) ? h[yy * WC + xx] : 0.f;
        };
        float sb = g(yf, xf) * (1.f - tx) * (1.f - ty)
                 + g(yf, xf + 1) * tx * (1.f - ty)
                 + g(yf + 1, xf) * (1.f - tx) * ty
                 + g(yf + 1, xf + 1) * tx * ty;
        float score = snear * sb;          // >= 0
        unsigned slot = atomicAdd(&cnt[b], 1u);
        if (slot < CAP)
          cand[(size_t)b * CAP + slot] =
              make_uint2(__float_as_uint(score), (unsigned)gx | ((unsigned)gy << 16));
      }
    }
  }
}

// ---------------------------------------------------------------------------
// Kernel 4: per-batch top-4096 by score. 12-bit histogram radix select for
// the threshold, LDS compaction, then 4096-wide bitonic sort (descending).
// ---------------------------------------------------------------------------
#define NBIN 4096
__global__ void k_select(const uint2* __restrict__ cand, const unsigned* __restrict__ cnt,
                         float* __restrict__ pos_out, float* __restrict__ score_out) {
  __shared__ unsigned hist[NBIN];
  __shared__ unsigned long long keys[TOPK];
  __shared__ unsigned scal[8];   // 0:above 1:need 2:aboveCtr 3:tbin 4:tieCtr
  int b = blockIdx.x, tid = threadIdx.x;
  unsigned n0 = cnt[b];
  unsigned n = (n0 < CAP) ? n0 : CAP;
  for (int i = tid; i < NBIN; i += blockDim.x) hist[i] = 0u;
  for (int i = tid; i < TOPK; i += blockDim.x) keys[i] = 0ull;
  if (tid < 8) scal[tid] = 0u;
  __syncthreads();
  const uint2* cd = cand + (size_t)b * CAP;
  for (unsigned i = tid; i < n; i += blockDim.x)
    atomicAdd(&hist[cd[i].x >> 20], 1u);     // positive floats: bits monotone
  __syncthreads();
  if (tid == 0) {
    unsigned acc = 0; int t = NBIN - 1;
    for (; t >= 0; --t) { if (acc + hist[t] >= TOPK) break; acc += hist[t]; }
    scal[0] = acc;
    scal[1] = (t >= 0) ? (TOPK - acc) : 0u;
    scal[3] = (unsigned)t;
  }
  __syncthreads();
  unsigned above = scal[0], need = scal[1];
  int tbin = (int)scal[3];
  for (unsigned i = tid; i < n; i += blockDim.x) {
    uint2 e = cd[i];
    int bin = (int)(e.x >> 20);
    unsigned long long kk = ((unsigned long long)e.x << 32) | (unsigned long long)e.y;
    if (bin > tbin) {
      unsigned s = atomicAdd(&scal[2], 1u);
      keys[s] = kk;
    } else if (bin == tbin) {
      unsigned s = atomicAdd(&scal[4], 1u);
      if (s < need) keys[above + s] = kk;
    }
  }
  __syncthreads();
  // bitonic sort, descending on (scoreBits, packedPos)
  for (unsigned k = 2; k <= TOPK; k <<= 1) {
    for (unsigned j = k >> 1; j > 0; j >>= 1) {
      for (unsigned t = tid; t < (TOPK / 2); t += blockDim.x) {
        unsigned i = ((t & ~(j - 1)) << 1) | (t & (j - 1));
        unsigned p = i | j;
        bool desc = ((i & k) == 0);
        unsigned long long a = keys[i], c = keys[p];
        if ((a < c) == desc) { keys[i] = c; keys[p] = a; }
      }
      __syncthreads();
    }
  }
  unsigned tot = (tbin >= 0) ? (unsigned)TOPK : above;
  for (int i = tid; i < TOPK; i += blockDim.x) {
    float sc; unsigned px, py;
    if ((unsigned)i < tot) {
      unsigned long long kk = keys[i];
      sc = __uint_as_float((unsigned)(kk >> 32));
      unsigned pp = (unsigned)kk;
      px = pp & 0xFFFFu; py = (pp >> 16) & 0xFFFFu;
    } else { sc = -1.f; px = 0u; py = 0u; }
    size_t o = (size_t)b * TOPK + i;
    pos_out[o * 2 + 0] = (float)px;
    pos_out[o * 2 + 1] = (float)py;
    score_out[o] = sc;
  }
}

// ---------------------------------------------------------------------------
// Kernel 5: bicubic sampling of per-pixel-normalized M1 at selected points,
// then L2-renormalize the 64-dim descriptor. One wave per keypoint; each
// lane owns 2 channels, norm via shuffle tree.
// ---------------------------------------------------------------------------
__device__ __forceinline__ void cubic_w(float t, float w[4]) {
  const float Ac = -0.75f;
  auto c1 = [&](float x) { return ((Ac + 2.f) * x - (Ac + 3.f)) * x * x + 1.f; };
  auto c2 = [&](float x) { return (((x - 5.f) * x + 8.f) * x - 4.f) * Ac; };
  w[0] = c2(t + 1.f); w[1] = c1(t); w[2] = c1(1.f - t); w[3] = c2(2.f - t);
}

__global__ void k_feats(const float* __restrict__ M1, const float* __restrict__ invn,
                        const float* __restrict__ pos_out, float* __restrict__ feats) {
  int kp = blockIdx.x * (blockDim.x >> 5) + (threadIdx.x >> 5);
  if (kp >= BATCH * TOPK) return;
  int lane = threadIdx.x & 31;
  int b = kp >> 12;
  float px = pos_out[(size_t)kp * 2 + 0];
  float py = pos_out[(size_t)kp * 2 + 1];
  float u = px * (160.f / 1279.f) - 0.5f;
  float v = py * (160.f / 1279.f) - 0.5f;
  float x0f = floorf(u), y0f = floorf(v);
  int x0 = (int)x0f, y0 = (int)y0f;
  float wx[4], wy[4];
  cubic_w(u - x0f, wx);
  cubic_w(v - y0f, wy);
  const float* m = M1 + (size_t)b * 64 * NPIX;
  const float* iv = invn + (size_t)b * NPIX;
  int c0 = lane, c1 = lane + 32;
  float a0 = 0.f, a1 = 0.f;
#pragma unroll
  for (int j = 0; j < 4; ++j) {
    int yy = y0 - 1 + j;
    if (yy < 0 || yy >= HC) continue;
#pragma unroll
    for (int i = 0; i < 4; ++i) {
      int xx = x0 - 1 + i;
      if (xx < 0 || xx >= WC) continue;
      float wgt = wy[j] * wx[i];
      size_t p = (size_t)yy * WC + xx;
      float s = iv[p] * wgt;
      a0 += s * m[(size_t)c0 * NPIX + p];
      a1 += s * m[(size_t)c1 * NPIX + p];
    }
  }
  float ss = a0 * a0 + a1 * a1;
#pragma unroll
  for (int o = 16; o > 0; o >>= 1) ss += __shfl_xor(ss, o, 32);
  float invN = 1.f / fmaxf(sqrtf(ss), 1e-12f);
  float* out = feats + (size_t)kp * 64;
  out[c0] = a0 * invN;
  out[c1] = a1 * invN;
}

// ---------------------------------------------------------------------------
extern "C" void kernel_launch(void* const* d_in, const int* in_sizes, int n_in,
                              void* d_out, int out_size, void* d_ws, size_t ws_size,
                              hipStream_t stream) {
  const float* M1 = (const float*)d_in[0];
  const float* K1 = (const float*)d_in[1];
  const float* H1 = (const float*)d_in[2];

  float* pos_out   = (float*)d_out;                       // 8*4096*2
  float* score_out = pos_out + (size_t)BATCH * TOPK * 2;  // 8*4096
  float* feats_out = score_out + (size_t)BATCH * TOPK;    // 8*4096*64

  char* ws = (char*)d_ws;
  unsigned* cnt  = (unsigned*)ws;                                   ws += 256;
  uint2*    cand = (uint2*)ws;        ws += (size_t)BATCH * CAP * sizeof(uint2);
  float*    invn = (float*)ws;        ws += (size_t)BATCH * NPIX * sizeof(float);
  float*    g1   = (float*)ws;        ws += GUARDF * sizeof(float);
  float*    K1h  = (float*)ws;        ws += (size_t)BATCH * HFULL * WFULL * sizeof(float);
  float*    g2   = (float*)ws;

  k_init<<<1, 256, 0, stream>>>(cnt, g1, g2);
  k_heatmap<<<(BATCH * NPIX + 255) / 256, 256, 0, stream>>>(K1, K1h);
  k_invnorm<<<(BATCH * NPIX / 16 + 7) / 8, 256, 0, stream>>>(M1, invn);
  k_nms<<<BATCH * 40 * 40, 256, 0, stream>>>(K1h, H1, cnt, cand);
  k_select<<<BATCH, 1024, 0, stream>>>(cand, cnt, pos_out, score_out);
  k_feats<<<(BATCH * TOPK + 7) / 8, 256, 0, stream>>>(M1, invn, pos_out, feats_out);
}
